// HisRepItself_23433341567726
// MI455X (gfx1250) — matrix operand order repaired and checked by reading, not compiled, not run on hardware
//
#include <hip/hip_runtime.h>
#include <hip/hip_bf16.h>
#include <math.h>

typedef _Float16 v16h __attribute__((ext_vector_type(16)));
typedef _Float16 h8   __attribute__((ext_vector_type(8)));
typedef float    v8f  __attribute__((ext_vector_type(8)));

#define BN_INV 0.999995000037f   // 1/sqrt(1+1e-5)

// ---------------- workspace layout (bytes) ----------------
static constexpr size_t OFF_DCT   = 0;                         // 10x20 f32
static constexpr size_t OFF_X     = 1024;                      // 128x66x20 f32
static constexpr size_t OFF_GC1WT = OFF_X     + 675840;        // 256x32 f16
static constexpr size_t OFF_GC7WT = OFF_GC1WT + 16384;         // 32x256 f16
static constexpr size_t OFF_W1T   = OFF_GC7WT + 16384;         // 12x256x256 f16
static constexpr size_t OFF_W2T   = OFF_W1T   + 1572864;       // 12x256x256 f16
static constexpr size_t OFF_A1    = OFF_W2T   + 1572864;       // 80x96 f16
static constexpr size_t OFF_AB1   = OFF_A1    + 15360;         // 12x80x96 f16
static constexpr size_t OFF_AB2   = OFF_AB1   + 184320;        // 12x80x96 f16
static constexpr size_t OFF_A7    = OFF_AB2   + 184320;        // 80x96 f16
// total ~4.26 MB

// ============================================================
// Kernel 0: precompute DCT matrix, f16-transposed weights,
//           zero-padded f16 att matrices (80 rows x 96 cols).
// ============================================================
__global__ __launch_bounds__(256)
void k_precompute(const float* __restrict__ gc1_w,  const float* __restrict__ gc1_att,
                  const float* __restrict__ gcb_w1, const float* __restrict__ gcb_att1,
                  const float* __restrict__ gcb_w2, const float* __restrict__ gcb_att2,
                  const float* __restrict__ gc7_w,  const float* __restrict__ gc7_att,
                  char* __restrict__ ws)
{
  float*    dct10 = (float*)(ws + OFF_DCT);
  _Float16* gc1wT = (_Float16*)(ws + OFF_GC1WT);
  _Float16* gc7wT = (_Float16*)(ws + OFF_GC7WT);
  _Float16* w1T   = (_Float16*)(ws + OFF_W1T);
  _Float16* w2T   = (_Float16*)(ws + OFF_W2T);
  _Float16* a1P   = (_Float16*)(ws + OFF_A1);
  _Float16* ab1P  = (_Float16*)(ws + OFF_AB1);
  _Float16* ab2P  = (_Float16*)(ws + OFF_AB2);
  _Float16* a7P   = (_Float16*)(ws + OFF_A7);

  int tid = blockIdx.x * blockDim.x + threadIdx.x;
  int nth = gridDim.x * blockDim.x;
  const float PI = 3.14159265358979323846f;

  // orthonormal DCT-II rows 0..9 (idct = dct^T)
  for (int i = tid; i < 200; i += nth) {
    int j = i / 20, t = i - j * 20;
    float w = (j == 0) ? sqrtf(1.0f / 20.0f) : sqrtf(2.0f / 20.0f);
    dct10[i] = w * cosf(PI * ((float)t + 0.5f) * (float)j / 20.0f);
  }
  // gc1_w (20,256) -> WT[out=256][kp=32]
  for (int i = tid; i < 256 * 32; i += nth) {
    int f = i >> 5, k = i & 31;
    gc1wT[i] = (k < 20) ? (_Float16)gc1_w[k * 256 + f] : (_Float16)0.0f;
  }
  // gc7_w (256,20) -> WT[out pad 32][k=256]
  for (int i = tid; i < 32 * 256; i += nth) {
    int o = i >> 8, k = i & 255;
    gc7wT[i] = (o < 20) ? (_Float16)gc7_w[k * 20 + o] : (_Float16)0.0f;
  }
  // gcb_w1/w2 (12,256,256) in->out  ->  WT[l][out][in]
  for (int i = tid; i < 12 * 65536; i += nth) {
    int l = i >> 16, r = i & 65535, o = r >> 8, k = r & 255;
    int s = (l << 16) + k * 256 + o;
    w1T[i] = (_Float16)gcb_w1[s];
    w2T[i] = (_Float16)gcb_w2[s];
  }
  // 26 att matrices (66x66) -> padded f16 [80][96], pads = 0
  for (int i = tid; i < 26 * 7680; i += nth) {
    int m = i / 7680, r = i - m * 7680;
    int n = r / 96, c = r - n * 96;
    const float* srcA; _Float16* dst;
    if (m == 0)       { srcA = gc1_att;                     dst = a1P; }
    else if (m < 13)  { srcA = gcb_att1 + (m - 1) * 4356;   dst = ab1P + (m - 1) * 7680; }
    else if (m < 25)  { srcA = gcb_att2 + (m - 13) * 4356;  dst = ab2P + (m - 13) * 7680; }
    else              { srcA = gc7_att;                     dst = a7P; }
    float v = (n < 66 && c < 66) ? srcA[n * 66 + c] : 0.0f;
    dst[r] = (_Float16)v;
  }
}

// ============================================================
// Kernel 1: attention front-end. One workgroup per batch.
// ============================================================
__global__ __launch_bounds__(256)
void k_front(const float* __restrict__ src,
             const float* __restrict__ qw1, const float* __restrict__ qw2,
             const float* __restrict__ kw1, const float* __restrict__ kw2,
             char* __restrict__ ws)
{
  const float* dct10 = (const float*)(ws + OFF_DCT);
  float*       xout  = (float*)(ws + OFF_X);
  int b = blockIdx.x, tid = threadIdx.x;

  extern __shared__ char smem_raw[];
  float* s_src = (float*)smem_raw;       // 50*66 = 3300
  float* s_c1  = s_src + 3300;           // 256*35
  float* s_key = s_c1  + 8960;           // 256*31
  float* s_q1  = s_key + 7936;           // 256*5
  float* s_q   = s_q1  + 1280;           // 256
  float* s_att = s_q   + 256;            // 32
  float* s_p   = s_att + 32;             // 66*20
  float* s_d   = s_p   + 1320;           // 200

  for (int i = tid; i < 3300; i += 256) s_src[i] = src[b * 3300 + i];
  for (int i = tid; i < 200;  i += 256) s_d[i]   = dct10[i];
  __syncthreads();

  // key conv1: (66,40)->(256,35), thread = out channel
  {
    int co = tid;
    float acc[35];
#pragma unroll
    for (int t = 0; t < 35; ++t) acc[t] = 0.0f;
    for (int f = 0; f < 66; ++f)
      for (int k = 0; k < 6; ++k) {
        float wv = kw1[co * 396 + f * 6 + k] * 1e-3f;
        const float* sp = s_src + k * 66 + f;
#pragma unroll
        for (int t = 0; t < 35; ++t) acc[t] = fmaf(wv, sp[t * 66], acc[t]);
      }
#pragma unroll
    for (int t = 0; t < 35; ++t) s_c1[co * 35 + t] = fmaxf(acc[t], 0.0f);
  }
  // query conv1: (66,10)->(256,5)
  {
    int co = tid;
    float acc[5];
#pragma unroll
    for (int t = 0; t < 5; ++t) acc[t] = 0.0f;
    for (int f = 0; f < 66; ++f)
      for (int k = 0; k < 6; ++k) {
        float wv = qw1[co * 396 + f * 6 + k] * 1e-3f;
        const float* sp = s_src + (40 + k) * 66 + f;
#pragma unroll
        for (int t = 0; t < 5; ++t) acc[t] = fmaf(wv, sp[t * 66], acc[t]);
      }
#pragma unroll
    for (int t = 0; t < 5; ++t) s_q1[co * 5 + t] = fmaxf(acc[t], 0.0f);
  }
  __syncthreads();

  // key conv2: (256,35)->(256,31)
  {
    int co = tid;
    float acc[31];
#pragma unroll
    for (int t = 0; t < 31; ++t) acc[t] = 0.0f;
    for (int ci = 0; ci < 256; ++ci)
#pragma unroll
      for (int k = 0; k < 5; ++k) {
        float wv = kw2[co * 1280 + ci * 5 + k];
        const float* sp = s_c1 + ci * 35 + k;
#pragma unroll
        for (int t = 0; t < 31; ++t) acc[t] = fmaf(wv, sp[t], acc[t]);
      }
#pragma unroll
    for (int t = 0; t < 31; ++t) s_key[co * 31 + t] = fmaxf(acc[t], 0.0f);
  }
  // query conv2: (256,5)->(256,1)
  {
    int co = tid;
    float a = 0.0f;
    for (int ci = 0; ci < 256; ++ci)
#pragma unroll
      for (int k = 0; k < 5; ++k)
        a = fmaf(qw2[co * 1280 + ci * 5 + k], s_q1[ci * 5 + k], a);
    s_q[co] = fmaxf(a, 0.0f);
  }
  __syncthreads();

  // score + normalize
  if (tid < 31) {
    float s = 1e-15f;
    for (int co = 0; co < 256; ++co) s = fmaf(s_q[co], s_key[co * 31 + tid], s);
    s_att[tid] = s;
  }
  __syncthreads();
  if (tid == 0) {
    float tot = 0.0f;
    for (int i = 0; i < 31; ++i) tot += s_att[i];
    float inv = 1.0f / tot;
    for (int i = 0; i < 31; ++i) s_att[i] *= inv;
  }
  __syncthreads();

  // p[f][t] = sum_i att[i]*src[i+t][f]
  for (int i = tid; i < 1320; i += 256) {
    int f = i / 20, t = i - f * 20;
    float s = 0.0f;
    for (int ii = 0; ii < 31; ++ii) s = fmaf(s_att[ii], s_src[(ii + t) * 66 + f], s);
    s_p[f * 20 + t] = s;
  }
  __syncthreads();

  // x[b][f][0..9]=dct_in, [10..19]=dct_att
  for (int i = tid; i < 660; i += 256) {
    int f = i / 10, j = i - f * 10;
    float din = 0.0f, datt = 0.0f;
#pragma unroll
    for (int t = 0; t < 20; ++t) {
      float d = s_d[j * 20 + t];
      int row = (t < 10) ? (40 + t) : 49;
      din  = fmaf(d, s_src[row * 66 + f], din);
      datt = fmaf(d, s_p[f * 20 + t], datt);
    }
    xout[b * 1320 + f * 20 + j]      = din;
    xout[b * 1320 + f * 20 + 10 + j] = datt;
  }
}

// ============================================================
// Kernel 2: full GCN stack with WMMA f16 (f32 accumulate).
// One workgroup (8 waves) per batch through all 26 layers.
// ============================================================
enum { M_GC1 = 0, M_HONLY = 1, M_RESID = 2, M_GC7 = 3 };

__device__ __forceinline__ v8f wmma_step(const _Float16* ap, const _Float16* bp, v8f c)
{
  // A frag: lane<16 -> K{k0..+7, k0+16..+23}; lane>=16 (ap pre-offset +8) -> K{+8..15,+24..31}
  // B frag: lane<16 -> K{k0..+7, k0+16..+23}; lane>=16 (bp pre-offset +8) -> K{+8..15,+24..31}
  union { v16h v; h8 p[2]; } ua, ub;
  ua.p[0] = *(const h8*)(ap);
  ua.p[1] = *(const h8*)(ap + 16);
  ub.p[0] = *(const h8*)(bp);
  ub.p[1] = *(const h8*)(bp + 16);
  return __builtin_amdgcn_wmma_f32_16x16x32_f16(false, ua.v, false, ub.v, (short)0, c,
                                                false, false);
}

// T = h(80 x K) @ W(K x N);  store transposed f16 into tT[n][96]
// wave_u is scalar (readfirstlane) so the tile loop is uniform: EXEC stays
// all-ones across the WMMAs (ISA requirement) and loop control is SALU-only.
template <int KITERS, int NTILES, int HSTR, int WSTR>
__device__ __forceinline__ void gemm_T(const _Float16* __restrict__ h,
                                       const _Float16* __restrict__ WT,
                                       _Float16* __restrict__ tT, int wave_u, int lane)
{
  int l15 = lane & 15, hi = lane >> 4;
  for (int tile = wave_u; tile < 5 * NTILES; tile += 8) {
    int mt = tile / NTILES, nt = tile - mt * NTILES;
    const _Float16* ap = h  + (mt * 16 + l15) * HSTR + (hi ? 8 : 0);
    const _Float16* bp = WT + (nt * 16 + l15) * WSTR + (hi ? 8 : 0);
    v8f c = {};
#pragma unroll
    for (int kk = 0; kk < KITERS; ++kk) {
      c = wmma_step(ap + kk * 32, bp + kk * 32, c);
    }
    h8 hv;
#pragma unroll
    for (int v = 0; v < 8; ++v) hv[v] = (_Float16)c[v];
    *(h8*)(tT + (nt * 16 + l15) * 96 + mt * 16 + hi * 8) = hv;   // transposed store
  }
}

// Y = attP(80x96) @ T(96xN) + fused epilogue
template <int MODE, int NTILES>
__device__ __forceinline__ void gemm_Y(const _Float16* __restrict__ attS,
                                       const _Float16* __restrict__ tT,
                                       const float* __restrict__ bias,
                                       const float* __restrict__ g,
                                       const float* __restrict__ bb,
                                       float* __restrict__ y32, _Float16* __restrict__ h16,
                                       float* __restrict__ xs, int wave_u, int lane)
{
  int l15 = lane & 15, hi = lane >> 4;
  for (int tile = wave_u; tile < 5 * NTILES; tile += 8) {
    int mt = tile / NTILES, nt = tile - mt * NTILES;
    int fcol = nt * 16 + l15;
    const _Float16* ap = attS + (mt * 16 + l15) * 96 + (hi ? 8 : 0);
    const _Float16* bp = tT + fcol * 96 + (hi ? 8 : 0);
    v8f c = {};
#pragma unroll
    for (int kk = 0; kk < 3; ++kk) {
      c = wmma_step(ap + kk * 32, bp + kk * 32, c);
    }
    int nbase = mt * 16 + hi * 8;
#pragma unroll
    for (int v = 0; v < 8; ++v) {
      int n = nbase + v;
      if (MODE == M_GC7) {
        if (n < 66 && fcol < 20) {
          xs[n * 20 + fcol] = c[v] + bias[fcol] + xs[n * 20 + fcol];
        }
      } else {
        int idx = n * 256 + fcol;
        if (n < 66) {
          float t = tanhf((c[v] + bias[fcol]) * (g[idx] * BN_INV) + bb[idx]);
          if (MODE == M_RESID) t += y32[idx];
          if (MODE != M_HONLY) y32[idx] = t;
          h16[idx] = (_Float16)t;
        } else {
          h16[idx] = (_Float16)0.0f;   // keep pad rows clean & finite
        }
      }
    }
  }
}

__device__ __forceinline__ void copy_att(const _Float16* __restrict__ srcA,
                                         _Float16* __restrict__ dst, int tid)
{
  const uint4* s = (const uint4*)srcA;
  uint4* d = (uint4*)dst;
  for (int i = tid; i < 960; i += 256) d[i] = s[i];  // 80*96 halves
}

__global__ __launch_bounds__(256)
void k_gcn(char* __restrict__ ws,
           const float* __restrict__ gc1_b, const float* __restrict__ bn1_g,
           const float* __restrict__ bn1_b,
           const float* __restrict__ gcb_b1, const float* __restrict__ gcb_bn1_g,
           const float* __restrict__ gcb_bn1_b,
           const float* __restrict__ gcb_b2, const float* __restrict__ gcb_bn2_g,
           const float* __restrict__ gcb_bn2_b,
           const float* __restrict__ gc7_b, float* __restrict__ out)
{
  const float*    dct10 = (const float*)(ws + OFF_DCT);
  const float*    xin   = (const float*)(ws + OFF_X);
  const _Float16* gc1wT = (const _Float16*)(ws + OFF_GC1WT);
  const _Float16* gc7wT = (const _Float16*)(ws + OFF_GC7WT);
  const _Float16* w1T   = (const _Float16*)(ws + OFF_W1T);
  const _Float16* w2T   = (const _Float16*)(ws + OFF_W2T);
  const _Float16* a1P   = (const _Float16*)(ws + OFF_A1);
  const _Float16* ab1P  = (const _Float16*)(ws + OFF_AB1);
  const _Float16* ab2P  = (const _Float16*)(ws + OFF_AB2);
  const _Float16* a7P   = (const _Float16*)(ws + OFF_A7);

  int b = blockIdx.x, tid = threadIdx.x;
  int lane = tid & 31;
  // wave index is uniform within a wave: force it into an SGPR so the tile
  // loops are scalar and EXEC remains all-ones around every WMMA.
  int wave_u = __builtin_amdgcn_readfirstlane(tid >> 5);

  extern __shared__ char smem_raw[];
  _Float16* h16  = (_Float16*)smem_raw;      // 80*256
  _Float16* tT   = h16  + 80 * 256;          // 256*96
  _Float16* attS = tT   + 256 * 96;          // 80*96
  float*    y32  = (float*)(attS + 80 * 96); // 66*256
  float*    xs   = y32 + 66 * 256;           // 66*20
  float*    sd   = xs  + 1320;               // 200

  for (int i = tid; i < 80 * 256; i += 256) h16[i] = (_Float16)0.0f;
  for (int i = tid; i < 256 * 96; i += 256) tT[i]  = (_Float16)0.0f;  // pads m>=80 stay 0
  for (int i = tid; i < 200; i += 256) sd[i] = dct10[i];
  __syncthreads();
  for (int i = tid; i < 1320; i += 256) {
    float v = xin[b * 1320 + i];
    xs[i] = v;
    int n = i / 20, k = i - n * 20;
    h16[n * 32 + k] = (_Float16)v;           // gc1 view: [80][32]
  }
  copy_att(a1P, attS, tid);
  __syncthreads();

  // gc1: x(66x20) @ W(20x256) -> att mix -> bn -> tanh
  gemm_T<1, 16, 32, 32>(h16, gc1wT, tT, wave_u, lane);
  __syncthreads();
  gemm_Y<M_GC1, 16>(attS, tT, gc1_b, bn1_g, bn1_b, y32, h16, nullptr, wave_u, lane);
  __syncthreads();

  // 12 residual blocks
  for (int l = 0; l < 12; ++l) {
    copy_att(ab1P + l * 7680, attS, tid);
    gemm_T<8, 16, 256, 256>(h16, w1T + l * 65536, tT, wave_u, lane);
    __syncthreads();
    gemm_Y<M_HONLY, 16>(attS, tT, gcb_b1 + l * 256, gcb_bn1_g + l * 16896,
                        gcb_bn1_b + l * 16896, y32, h16, nullptr, wave_u, lane);
    __syncthreads();
    copy_att(ab2P + l * 7680, attS, tid);
    gemm_T<8, 16, 256, 256>(h16, w2T + l * 65536, tT, wave_u, lane);
    __syncthreads();
    gemm_Y<M_RESID, 16>(attS, tT, gcb_b2 + l * 256, gcb_bn2_g + l * 16896,
                        gcb_bn2_b + l * 16896, y32, h16, nullptr, wave_u, lane);
    __syncthreads();
  }

  // gc7: y @ W(256x20) -> att mix -> +b -> +x   (into xs)
  copy_att(a7P, attS, tid);
  gemm_T<8, 2, 256, 256>(h16, gc7wT, tT, wave_u, lane);
  __syncthreads();
  gemm_Y<M_GC7, 2>(attS, tT, gc7_b, nullptr, nullptr, nullptr, nullptr, xs, wave_u, lane);
  __syncthreads();

  // out[o][f] = sum_j idct[o][j] * y[f][j],  idct = dct^T
  for (int i = tid; i < 1320; i += 256) {
    int o = i / 66, f = i - o * 66;
    float s = 0.0f;
#pragma unroll
    for (int j = 0; j < 10; ++j) s = fmaf(sd[j * 20 + o], xs[f * 20 + j], s);
    out[b * 1320 + o * 66 + f] = s;
  }
}

// ============================================================
extern "C" void kernel_launch(void* const* d_in, const int* in_sizes, int n_in,
                              void* d_out, int out_size, void* d_ws, size_t ws_size,
                              hipStream_t stream)
{
  const float* src      = (const float*)d_in[0];
  const float* qw1      = (const float*)d_in[1];
  const float* qw2      = (const float*)d_in[2];
  const float* kw1      = (const float*)d_in[3];
  const float* kw2      = (const float*)d_in[4];
  const float* gc1_w    = (const float*)d_in[5];
  const float* gc1_att  = (const float*)d_in[6];
  const float* gc1_b    = (const float*)d_in[7];
  const float* bn1_g    = (const float*)d_in[8];
  const float* bn1_b    = (const float*)d_in[9];
  const float* gcb_w1   = (const float*)d_in[10];
  const float* gcb_att1 = (const float*)d_in[11];
  const float* gcb_b1   = (const float*)d_in[12];
  const float* gcb_bn1g = (const float*)d_in[13];
  const float* gcb_bn1b = (const float*)d_in[14];
  const float* gcb_w2   = (const float*)d_in[15];
  const float* gcb_att2 = (const float*)d_in[16];
  const float* gcb_b2   = (const float*)d_in[17];
  const float* gcb_bn2g = (const float*)d_in[18];
  const float* gcb_bn2b = (const float*)d_in[19];
  const float* gc7_w    = (const float*)d_in[20];
  const float* gc7_att  = (const float*)d_in[21];
  const float* gc7_b    = (const float*)d_in[22];
  char*  ws  = (char*)d_ws;
  float* out = (float*)d_out;

  k_precompute<<<512, 256, 0, stream>>>(gc1_w, gc1_att, gcb_w1, gcb_att1,
                                        gcb_w2, gcb_att2, gc7_w, gc7_att, ws);

  size_t sm1 = (size_t)23284 * sizeof(float);   // ~93 KB
  k_front<<<128, 256, sm1, stream>>>(src, qw1, qw2, kw1, kw2, ws);

  size_t sm2 = (size_t)(80 * 256 + 256 * 96 + 80 * 96) * 2 +
               (size_t)(66 * 256 + 1320 + 200) * 4;   // ~179 KB (<320KB/WGP)
  k_gcn<<<128, 256, sm2, stream>>>(ws, gc1_b, bn1_g, bn1_b,
                                   gcb_b1, gcb_bn1g, gcb_bn1b,
                                   gcb_b2, gcb_bn2g, gcb_bn2b,
                                   gc7_b, out);
}
